// DeltaNet_22488448762170
// MI455X (gfx1250) — matrix operand order repaired and checked by reading, not compiled, not run on hardware
//
#include <hip/hip_runtime.h>
#include <hip/hip_bf16.h>
#include <stdint.h>

// ---------------- CDNA5 WMMA plumbing ----------------
typedef __bf16 bf16x16 __attribute__((ext_vector_type(16)));
typedef float  f32x8   __attribute__((ext_vector_type(8)));

#define B_   4
#define T_   2048
#define HID_ 1024
#define H_   8
#define J_   2
#define D_   128
#define BT_  (B_ * T_)
#define CHUNK 64

__device__ __forceinline__ uint16_t f2bf(float f) {
  union { float f; uint32_t u; } x; x.f = f;
  uint32_t u = x.u;
  u += 0x7FFFu + ((u >> 16) & 1u);     // round-to-nearest-even
  return (uint16_t)(u >> 16);
}
__device__ __forceinline__ float bf2f(uint16_t h) {
  union { uint32_t u; float f; } x; x.u = ((uint32_t)h) << 16;
  return x.f;
}
__device__ __forceinline__ uint32_t pack2bf(float a, float b) {
  return (uint32_t)f2bf(a) | ((uint32_t)f2bf(b) << 16);
}

union FragU { uint32_t u[8]; bf16x16 v; };

// A-matrix (16x32 bf16) fragment from row-major [m][k] bf16 LDS tile (stride rsU32 u32).
// ISA layout: lanes 0-15 M=lane; vgpr i<4 -> K = half*8 + 2i; i>=4 -> K = 16 + half*8 + 2(i-4).
__device__ __forceinline__ bf16x16 load_a_frag(const uint32_t* lds, int rsU32,
                                               int m0, int k0, int lane) {
  int m = m0 + (lane & 15);
  int half = (lane >> 4) & 1;
  const uint32_t* row = lds + m * rsU32 + (k0 >> 1);
  FragU r;
#pragma unroll
  for (int i = 0; i < 8; ++i)
    r.u[i] = row[((i >= 4) ? 8 : 0) + half * 4 + (i & 3)];
  return r.v;
}

// B-matrix (32x16 bf16, B[k][n]) fragment from LDS stored TRANSPOSED as [n][k] row-major.
// Layout (matches A): lanes 0-15 N=lane; vgpr i<4 -> K = half*8 + 2i; i>=4 -> K = 16 + half*8 + 2(i-4).
__device__ __forceinline__ bf16x16 load_b_frag(const uint32_t* lds, int rsU32,
                                               int k0, int n0, int lane) {
  int n = n0 + (lane & 15);
  int half = (lane >> 4) & 1;
  const uint32_t* row = lds + n * rsU32 + (k0 >> 1);
  FragU r;
#pragma unroll
  for (int i = 0; i < 8; ++i) r.u[i] = row[((i >= 4) ? 8 : 0) + half * 4 + (i & 3)];
  return r.v;
}

__device__ __forceinline__ f32x8 wmma_bf16(bf16x16 a, bf16x16 b, f32x8 c) {
  return __builtin_amdgcn_wmma_f32_16x16x32_bf16(false, a, false, b, (short)0, c, false, false);
}

// ---------------- CDNA5 async global->LDS staging (ASYNCcnt path) ----------------
// VDST VGPR carries the LDS byte offset; on AMDGCN the low 32 bits of a generic
// pointer to a __shared__ object are exactly that offset.
__device__ __forceinline__ uint32_t lds_off32(const void* p) {
  return (uint32_t)(uintptr_t)p;
}
__device__ __forceinline__ void async_load_b128(uint32_t lds_off, const void* gaddr) {
  asm volatile("global_load_async_to_lds_b128 %0, %1, off"
               :: "v"(lds_off), "v"(gaddr) : "memory");
}
__device__ __forceinline__ void wait_async0() {
  asm volatile("s_wait_asynccnt 0" ::: "memory");
}

// ---------------- elementwise helpers ----------------
__global__ void k_cvt_x(const float* __restrict__ x, uint16_t* __restrict__ xbf, int n) {
  int i = blockIdx.x * 256 + threadIdx.x;
  if (i < n) xbf[i] = f2bf(x[i]);
}

__global__ void k_pack_w(const float* __restrict__ Wq, const float* __restrict__ Wk,
                         const float* __restrict__ Wv, const float* __restrict__ Wout,
                         uint16_t* __restrict__ Wcat, uint16_t* __restrict__ Woutb) {
  int i = blockIdx.x * 256 + threadIdx.x;
  const int NW = 1024 * 1024;
  if (i < NW)              Wcat[i] = f2bf(Wq[i]);
  else if (i < 2 * NW)     Wcat[i] = f2bf(Wk[i - NW]);
  else if (i < 3 * NW)     Wcat[i] = f2bf(Wv[i - 2 * NW]);
  else if (i < 4 * NW)     Woutb[i - 3 * NW] = f2bf(Wout[i - 3 * NW]);
}

// ---------------- fused QKV projection GEMM ----------------
// out[m,n] = sum_k x[m,k] * Wcat[n,k]; n<1024 -> q (elu+1), <2048 -> k (elu+1), else v.
// Tiles staged with GLOBAL_LOAD_ASYNC_TO_LDS_B128; each wave drains its own
// ASYNCcnt before the barrier (barriers do not cover async copies).
__global__ __launch_bounds__(256) void k_proj_gemm(
    const uint32_t* __restrict__ xbf, const uint32_t* __restrict__ wbf,
    float* __restrict__ qf, float* __restrict__ kf, float* __restrict__ vf) {
  __shared__ uint32_t At[64 * 17];
  __shared__ uint32_t Bt[64 * 17];
  int m0 = blockIdx.x * 64, n0 = blockIdx.y * 64;
  int tid = threadIdx.x, lane = tid & 31, w = tid >> 5;
  int wm = w >> 1, wn = w & 1;
  f32x8 acc0 = {}, acc1 = {};
  int r = tid >> 2, c4 = (tid & 3) * 4;
  uint32_t ldsA = lds_off32(&At[r * 17 + c4]);
  uint32_t ldsB = lds_off32(&Bt[r * 17 + c4]);
  for (int kk = 0; kk < 1024; kk += 32) {
    __syncthreads();
    async_load_b128(ldsA, xbf + (m0 + r) * 512 + (kk >> 1) + c4);
    async_load_b128(ldsB, wbf + (n0 + r) * 512 + (kk >> 1) + c4);
    wait_async0();
    __syncthreads();
    bf16x16 a  = load_a_frag(At, 17, wm * 16, 0, lane);
    bf16x16 b0 = load_b_frag(Bt, 17, 0, wn * 32, lane);
    bf16x16 b1 = load_b_frag(Bt, 17, 0, wn * 32 + 16, lane);
    acc0 = wmma_bf16(a, b0, acc0);
    acc1 = wmma_bf16(a, b1, acc1);
  }
  int half = (lane >> 4) & 1;
  int mBase = m0 + wm * 16 + half * 8;
  int nBase = n0 + wn * 32 + (lane & 15);
#pragma unroll
  for (int i = 0; i < 8; ++i) {
    int m = mBase + i;
    int b = m >> 11, t = m & 2047;
#pragma unroll
    for (int tt = 0; tt < 2; ++tt) {
      int n = nBase + tt * 16;
      float v = (tt == 0) ? acc0[i] : acc1[i];
      int which = n >> 10;
      int col = n & 1023;
      int h = col >> 7, d = col & 127;
      size_t dst = (((size_t)(b * H_ + h)) * T_ + t) * D_ + d;
      if (which == 0)      qf[dst] = (v > 0.f) ? v + 1.f : __expf(v);
      else if (which == 1) kf[dst] = (v > 0.f) ? v + 1.f : __expf(v);
      else                 vf[dst] = v;
    }
  }
}

// ---------------- small projections + nonlinearities ----------------
// Per token: beta_eff = min(sigmoid(beta)*r_h, 0.9999), w = softmax(mix) over J=2, gate = sigmoid.
__global__ __launch_bounds__(256) void k_small_proj(
    const float* __restrict__ x, const float* __restrict__ Wbeta, const float* __restrict__ bbeta,
    const float* __restrict__ Wmix, const float* __restrict__ bmix,
    const float* __restrict__ Wgate, const float* __restrict__ bgate,
    float* __restrict__ betaeff, float* __restrict__ wsoft, float* __restrict__ gate) {
  __shared__ float res[40];
  int tok = blockIdx.x;
  int b = tok >> 11, t = tok & 2047;
  const float* xr = x + (size_t)tok * HID_;
  int tid = threadIdx.x, lane = tid & 31, w = tid >> 5;
  for (int o = w; o < 40; o += 8) {
    const float* wr = (o < 16) ? (Wbeta + o * HID_)
                    : (o < 32) ? (Wmix + (o - 16) * HID_)
                               : (Wgate + (o - 32) * HID_);
    float s = 0.f;
    for (int k = lane; k < HID_; k += 32) s += xr[k] * wr[k];
#pragma unroll
    for (int off = 16; off > 0; off >>= 1) s += __shfl_xor(s, off, 32);
    if (lane == 0) res[o] = s;
  }
  __syncthreads();
  if (tid < H_) {
    int h = tid;
    float r = __expf(-exp2f(-(float)(h + 1)));   // exp(-alibi_slope_h), slopes=0.5^(h+1)
    size_t bh = (size_t)b * H_ + h;
#pragma unroll
    for (int jj = 0; jj < J_; ++jj) {
      float be = res[h * 2 + jj] + bbeta[h * 2 + jj];
      float sg = 1.f / (1.f + __expf(-be));
      betaeff[(bh * J_ + jj) * T_ + t] = fminf(sg * r, 0.9999f);
    }
    float m0v = res[16 + h * 2 + 0] + bmix[h * 2 + 0];
    float m1v = res[16 + h * 2 + 1] + bmix[h * 2 + 1];
    float w0 = 1.f / (1.f + __expf(m1v - m0v));
    wsoft[(bh * J_ + 0) * T_ + t] = w0;
    wsoft[(bh * J_ + 1) * T_ + t] = 1.f - w0;
    float gv = res[32 + h] + bgate[h];
    gate[bh * T_ + t] = 1.f / (1.f + __expf(-gv));
  }
}

// ---------------- chunked decay-scan, all heavy math on WMMA ----------------
// One workgroup per (b,h,j). State S (128x128 f32, 64KB) lives in LDS for the
// whole T loop; a bf16 transposed copy feeds the q~@S WMMA GEMM. ~192KB LDS,
// legal on CDNA5 (320KB/workgroup).
__global__ __launch_bounds__(256) void k_scan(
    const float* __restrict__ qf, const float* __restrict__ kf, const float* __restrict__ vf,
    const float* __restrict__ betaeff, const float* __restrict__ wsoft,
    const float* __restrict__ gate, float* __restrict__ ybuf) {
  extern __shared__ unsigned char smem[];
  float*    S    = (float*)smem;                 // 128*128 f32
  uint32_t* SbfT = (uint32_t*)(S + 128 * 128);   // S^T bf16, [e][d], stride 65 u32
  uint32_t* qtb  = SbfT + 128 * 65;              // q~ bf16 [t][d], stride 65
  uint32_t* khb  = qtb + 64 * 65;                // k^ bf16 [s][d], stride 65
  uint32_t* khT  = khb + 64 * 65;                // k^T bf16 [d][s], stride 33
  uint32_t* vT   = khT + 128 * 33;               // v^T bf16 [e][s], stride 33
  float*    Af   = (float*)(vT + 128 * 33);      // A f32 [t][s], stride 65
  uint32_t* Amb  = (uint32_t*)(Af + 64 * 65);    // masked A bf16 [t][s], stride 33
  float*    cA   = (float*)(Amb + 64 * 33);      // 64 cumprod decays
  float*    skA  = cA + 64;                      // 64 g/c scales
  float*    wA   = skA + 64;                     // 64 softmax weights
  float*    den  = wA + 64;                      // 64 denominators
  float*    zv   = den + 64;                     // 128 z state

  int blk = blockIdx.x;
  int j = blk & 1;
  int bh = blk >> 1;
  int tid = threadIdx.x, lane = tid & 31, w = tid >> 5;

  const float* qp = qf + (size_t)bh * T_ * D_;
  const float* kp = kf + (size_t)bh * T_ * D_;
  const float* vp = vf + (size_t)bh * T_ * D_;
  const float* bp = betaeff + ((size_t)bh * J_ + j) * T_;
  const float* wp = wsoft   + ((size_t)bh * J_ + j) * T_;
  const float* gp = gate + (size_t)bh * T_;
  float* yp = ybuf + ((size_t)j * (B_ * H_) + bh) * T_ * D_;

  uint16_t* khT_h  = (uint16_t*)khT;   // stride 66 bf16
  uint16_t* vT_h   = (uint16_t*)vT;    // stride 66
  uint16_t* qtb_h  = (uint16_t*)qtb;   // stride 130
  uint16_t* Amb_h  = (uint16_t*)Amb;   // stride 66
  uint16_t* SbfT_h = (uint16_t*)SbfT;  // stride 130

  for (int i = tid; i < 128 * 128; i += 256) S[i] = 0.f;
  for (int i = tid; i < 128 * 65; i += 256) SbfT[i] = 0u;
  if (tid < 128) zv[tid] = 0.f;
  __syncthreads();

  for (int chunk = 0; chunk < T_ / CHUNK; ++chunk) {
    int t0 = chunk * CHUNK;
    if (tid < CHUNK) {
      cA[tid]  = bp[t0 + tid];   // beta (temp)
      skA[tid] = gp[t0 + tid];   // g (temp)
      wA[tid]  = wp[t0 + tid];
    }
    __syncthreads();
    if (tid == 0) {              // serial cumprod (64 steps; the ONLY serial math)
      float c = 1.f;
      for (int t = 0; t < CHUNK; ++t) { c *= cA[t]; cA[t] = c; }
    }
    __syncthreads();
    if (tid < CHUNK) skA[tid] = skA[tid] / cA[tid];
    __syncthreads();
    // build q~ = c_t*q, k^ = (g/c)*k, plus transposed bf16 copies of k^ and v
    for (int idx = tid; idx < CHUNK * 64; idx += 256) {
      int t = idx >> 6, p = idx & 63, d0 = p * 2;
      float ct = cA[t], sk = skA[t];
      const float* qr = qp + (size_t)(t0 + t) * D_;
      const float* kr = kp + (size_t)(t0 + t) * D_;
      const float* vr = vp + (size_t)(t0 + t) * D_;
      float q0 = qr[d0], q1 = qr[d0 + 1];
      float k0 = kr[d0], k1 = kr[d0 + 1];
      float v0 = vr[d0], v1 = vr[d0 + 1];
      qtb[t * 65 + p] = pack2bf(ct * q0, ct * q1);
      float kh0 = sk * k0, kh1 = sk * k1;
      khb[t * 65 + p] = pack2bf(kh0, kh1);
      khT_h[d0 * 66 + t] = f2bf(kh0);
      khT_h[(d0 + 1) * 66 + t] = f2bf(kh1);
      vT_h[d0 * 66 + t] = f2bf(v0);
      vT_h[(d0 + 1) * 66 + t] = f2bf(v1);
    }
    __syncthreads();
    // Phase A: A = q~ @ k^T  (64x64, K=128)
    {
      int wm = w >> 1, wn = w & 1;
      f32x8 a0 = {}, a1 = {};
      for (int kk = 0; kk < 128; kk += 32) {
        bf16x16 af = load_a_frag(qtb, 65, wm * 16, kk, lane);
        bf16x16 b0 = load_b_frag(khb, 65, kk, wn * 32, lane);
        bf16x16 b1 = load_b_frag(khb, 65, kk, wn * 32 + 16, lane);
        a0 = wmma_bf16(af, b0, a0);
        a1 = wmma_bf16(af, b1, a1);
      }
      int half = (lane >> 4) & 1;
      int mB = wm * 16 + half * 8;
      int nB = wn * 32 + (lane & 15);
#pragma unroll
      for (int i = 0; i < 8; ++i) {
        Af[(mB + i) * 65 + nB] = a0[i];
        Af[(mB + i) * 65 + nB + 16] = a1[i];
      }
    }
    __syncthreads();
    // causal mask -> bf16 A
    for (int idx = tid; idx < CHUNK * CHUNK; idx += 256) {
      int t = idx >> 6, s = idx & 63;
      Amb_h[t * 66 + s] = f2bf((s <= t) ? Af[t * 65 + s] : 0.f);
    }
    __syncthreads();
    // denom_t = rowsum(A_mask) + q~_t . z_prev + EPS
    if (tid < CHUNK) {
      int t = tid;
      float s = 1e-6f;
      for (int ss = 0; ss <= t; ++ss) s += Af[t * 65 + ss];
      for (int d = 0; d < D_; ++d) s += bf2f(qtb_h[t * 130 + d]) * zv[d];
      den[t] = s;
    }
    __syncthreads();
    // Phase B: num = q~ @ S_prev + A_mask @ V  (64x128); y = w*num/den
    {
      int wm = w >> 1, wn = w & 1;
      f32x8 acc[4] = {};
      for (int kk = 0; kk < 128; kk += 32) {
        bf16x16 af = load_a_frag(qtb, 65, wm * 16, kk, lane);
#pragma unroll
        for (int nt = 0; nt < 4; ++nt) {
          bf16x16 bb = load_b_frag(SbfT, 65, kk, wn * 64 + nt * 16, lane);
          acc[nt] = wmma_bf16(af, bb, acc[nt]);
        }
      }
      for (int kk = 0; kk < 64; kk += 32) {
        bf16x16 af = load_a_frag(Amb, 33, wm * 16, kk, lane);
#pragma unroll
        for (int nt = 0; nt < 4; ++nt) {
          bf16x16 bb = load_b_frag(vT, 33, kk, wn * 64 + nt * 16, lane);
          acc[nt] = wmma_bf16(af, bb, acc[nt]);
        }
      }
      int half = (lane >> 4) & 1;
      int mB = wm * 16 + half * 8;
      int nB = wn * 64 + (lane & 15);
#pragma unroll
      for (int nt = 0; nt < 4; ++nt)
#pragma unroll
        for (int i = 0; i < 8; ++i) {
          int t = mB + i, e = nB + nt * 16;
          yp[(size_t)(t0 + t) * D_ + e] = wA[t] * acc[nt][i] / den[t];
        }
    }
    __syncthreads();
    // Phase C: S = c_last*(S + k^T @ V)  (128x128, K=64); refresh bf16 S^T; z update
    {
      float cLast = cA[CHUNK - 1];
      f32x8 acc[8] = {};
      for (int kk = 0; kk < 64; kk += 32) {
        bf16x16 af = load_a_frag(khT, 33, w * 16, kk, lane);
#pragma unroll
        for (int nt = 0; nt < 8; ++nt) {
          bf16x16 bb = load_b_frag(vT, 33, kk, nt * 16, lane);
          acc[nt] = wmma_bf16(af, bb, acc[nt]);
        }
      }
      int half = (lane >> 4) & 1;
      int mB = w * 16 + half * 8;
      int nL = lane & 15;
#pragma unroll
      for (int nt = 0; nt < 8; ++nt)
#pragma unroll
        for (int i = 0; i < 8; ++i) {
          int m = mB + i, n = nt * 16 + nL;
          float sNew = cLast * (S[m * 128 + n] + acc[nt][i]);
          S[m * 128 + n] = sNew;
          SbfT_h[n * 130 + m] = f2bf(sNew);
        }
      __syncthreads();
      if (tid < 128) {
        float s = 0.f;
        for (int ss = 0; ss < CHUNK; ++ss) s += bf2f(khT_h[tid * 66 + ss]);
        zv[tid] = cLast * (zv[tid] + s);
      }
    }
    __syncthreads();
  }
}

// combine j=0/1 contributions, relayout [B,H,T,D] -> [B,T,HID], cast bf16
__global__ void k_comb(const float* __restrict__ y0, const float* __restrict__ y1,
                       uint16_t* __restrict__ ybf) {
  int i = blockIdx.x * 256 + threadIdx.x;
  if (i >= B_ * H_ * T_ * D_) return;
  int d = i & 127, t = (i >> 7) & 2047, h = (i >> 18) & 7, b = i >> 21;
  float v = y0[i] + y1[i];
  ybf[(((size_t)b * T_ + t) * HID_) + h * D_ + d] = f2bf(v);
}

// ---------------- output GEMM: out = y @ Wout^T + bout ----------------
__global__ __launch_bounds__(256) void k_out_gemm(
    const uint32_t* __restrict__ ybf, const uint32_t* __restrict__ wbf,
    const float* __restrict__ bout, float* __restrict__ outf) {
  __shared__ uint32_t At[64 * 17];
  __shared__ uint32_t Bt[64 * 17];
  int m0 = blockIdx.x * 64, n0 = blockIdx.y * 64;
  int tid = threadIdx.x, lane = tid & 31, w = tid >> 5;
  int wm = w >> 1, wn = w & 1;
  f32x8 acc0 = {}, acc1 = {};
  int r = tid >> 2, c4 = (tid & 3) * 4;
  uint32_t ldsA = lds_off32(&At[r * 17 + c4]);
  uint32_t ldsB = lds_off32(&Bt[r * 17 + c4]);
  for (int kk = 0; kk < 1024; kk += 32) {
    __syncthreads();
    async_load_b128(ldsA, ybf + (m0 + r) * 512 + (kk >> 1) + c4);
    async_load_b128(ldsB, wbf + (n0 + r) * 512 + (kk >> 1) + c4);
    wait_async0();
    __syncthreads();
    bf16x16 a  = load_a_frag(At, 17, wm * 16, 0, lane);
    bf16x16 b0 = load_b_frag(Bt, 17, 0, wn * 32, lane);
    bf16x16 b1 = load_b_frag(Bt, 17, 0, wn * 32 + 16, lane);
    acc0 = wmma_bf16(a, b0, acc0);
    acc1 = wmma_bf16(a, b1, acc1);
  }
  int half = (lane >> 4) & 1;
  int mBase = m0 + wm * 16 + half * 8;
  int nBase = n0 + wn * 32 + (lane & 15);
#pragma unroll
  for (int i = 0; i < 8; ++i) {
    int m = mBase + i;
    outf[(size_t)m * 1024 + nBase] = acc0[i] + bout[nBase];
    outf[(size_t)m * 1024 + nBase + 16] = acc1[i] + bout[nBase + 16];
  }
}

// ---------------- residual + LayerNorm ----------------
__global__ __launch_bounds__(256) void k_ln(const float* __restrict__ x,
                                            const float* __restrict__ outp,
                                            const float* __restrict__ g,
                                            const float* __restrict__ bb,
                                            float* __restrict__ out) {
  __shared__ float hbuf[1024];
  __shared__ float red[256];
  int tok = blockIdx.x;
  const float* xr = x + (size_t)tok * 1024;
  const float* orr = outp + (size_t)tok * 1024;
  int tid = threadIdx.x;
  float s = 0.f;
  for (int i = tid; i < 1024; i += 256) { float hv = xr[i] + orr[i]; hbuf[i] = hv; s += hv; }
  red[tid] = s; __syncthreads();
  for (int off = 128; off > 0; off >>= 1) { if (tid < off) red[tid] += red[tid + off]; __syncthreads(); }
  float mu = red[0] * (1.f / 1024.f);
  __syncthreads();
  float s2 = 0.f;
  for (int i = tid; i < 1024; i += 256) { float dv = hbuf[i] - mu; s2 += dv * dv; }
  red[tid] = s2; __syncthreads();
  for (int off = 128; off > 0; off >>= 1) { if (tid < off) red[tid] += red[tid + off]; __syncthreads(); }
  float inv = rsqrtf(red[0] * (1.f / 1024.f) + 1e-5f);
  for (int i = tid; i < 1024; i += 256)
    out[(size_t)tok * 1024 + i] = (hbuf[i] - mu) * inv * g[i] + bb[i];
}

// ---------------- host orchestration ----------------
extern "C" void kernel_launch(void* const* d_in, const int* in_sizes, int n_in,
                              void* d_out, int out_size, void* d_ws, size_t ws_size,
                              hipStream_t stream) {
  const float* x     = (const float*)d_in[0];
  const float* Wq    = (const float*)d_in[1];
  const float* Wk    = (const float*)d_in[2];
  const float* Wv    = (const float*)d_in[3];
  const float* Wbeta = (const float*)d_in[4];
  const float* bbeta = (const float*)d_in[5];
  const float* Wmix  = (const float*)d_in[6];
  const float* bmix  = (const float*)d_in[7];
  const float* Wgate = (const float*)d_in[8];
  const float* bgate = (const float*)d_in[9];
  const float* Wout  = (const float*)d_in[10];
  const float* bout  = (const float*)d_in[11];
  const float* ln_g  = (const float*)d_in[12];
  const float* ln_b  = (const float*)d_in[13];

  char* ws = (char*)d_ws;
  size_t off = 0;
  auto carve = [&](size_t bytes) { void* p = ws + off; off += (bytes + 255) & ~(size_t)255; return p; };

  uint16_t* xbf   = (uint16_t*)carve((size_t)BT_ * HID_ * 2);          // 16MB
  uint16_t* Wcat  = (uint16_t*)carve((size_t)3072 * 1024 * 2);         // 6MB
  uint16_t* Woutb = (uint16_t*)carve((size_t)1024 * 1024 * 2);         // 2MB
  float* qf       = (float*)carve((size_t)BT_ * HID_ * 4);             // 32MB [B,H,T,D]
  float* kf       = (float*)carve((size_t)BT_ * HID_ * 4);             // 32MB
  float* vf       = (float*)carve((size_t)BT_ * HID_ * 4);             // 32MB
  float* betaeff  = (float*)carve((size_t)B_ * H_ * J_ * T_ * 4);
  float* wsoft    = (float*)carve((size_t)B_ * H_ * J_ * T_ * 4);
  float* gateb    = (float*)carve((size_t)B_ * H_ * T_ * 4);
  float* ybuf     = (float*)carve((size_t)J_ * BT_ * HID_ * 4);        // 64MB (j=0,1)
  uint16_t* ybf   = (uint16_t*)carve((size_t)BT_ * HID_ * 2);          // 16MB
  float* outf     = qf;  // alias: q no longer needed after the scan

  const int NTOK = BT_ * HID_;                 // 8388608
  k_cvt_x<<<(NTOK + 255) / 256, 256, 0, stream>>>(x, xbf, NTOK);
  k_pack_w<<<(4 * 1024 * 1024 + 255) / 256, 256, 0, stream>>>(Wq, Wk, Wv, Wout, Wcat, Woutb);

  dim3 g1(BT_ / 64, 3072 / 64);
  k_proj_gemm<<<g1, 256, 0, stream>>>((const uint32_t*)xbf, (const uint32_t*)Wcat, qf, kf, vf);

  k_small_proj<<<BT_, 256, 0, stream>>>(x, Wbeta, bbeta, Wmix, bmix, Wgate, bgate,
                                        betaeff, wsoft, gateb);

  // dynamic LDS: S(64K) + SbfT(33.3K) + qtb/khb(2x16.6K) + khT/vT(2x16.9K)
  //            + Af(16.6K) + Amb(8.4K) + scalars(1.5K) = 192512 B (< 320KB/WG on CDNA5)
  size_t smem = (size_t)(128 * 128) * 4 + (size_t)(128 * 65) * 4
              + (size_t)(64 * 65) * 4 * 2 + (size_t)(128 * 33) * 4 * 2
              + (size_t)(64 * 65) * 4 + (size_t)(64 * 33) * 4
              + (size_t)(4 * 64 + 128) * 4;
  k_scan<<<B_ * H_ * J_, 256, smem, stream>>>(qf, kf, vf, betaeff, wsoft, gateb, ybuf);

  k_comb<<<(NTOK + 255) / 256, 256, 0, stream>>>(ybuf, ybuf + (size_t)BT_ * HID_, ybf);

  dim3 g2(BT_ / 64, 1024 / 64);
  k_out_gemm<<<g2, 256, 0, stream>>>((const uint32_t*)ybf, (const uint32_t*)Woutb, bout, outf);

  k_ln<<<BT_, 256, 0, stream>>>(x, outf, ln_g, ln_b, (float*)d_out);
}
